// PABlock_60825326846212
// MI455X (gfx1250) — matrix operand order repaired and checked by reading, not compiled, not run on hardware
//
#include <hip/hip_runtime.h>

#define NTOT 4096
#define CC   512
#define CRED 64
#define MTOT 640   // 64 (q) + 64 (k) + 512 (v) stacked output channels

typedef __bf16 bf16;
typedef __attribute__((ext_vector_type(16))) __bf16 v16bf;
typedef __attribute__((ext_vector_type(8)))  __bf16 v8bf;
typedef __attribute__((ext_vector_type(8)))  float  v8f;

#if defined(__has_builtin)
#if __has_builtin(__builtin_amdgcn_tensor_load_to_lds)
#define HAVE_TDM 1
#endif
#endif

static __device__ __forceinline__ bf16 f2bf(float f) {
    union { float f; unsigned u; } a; a.f = f;
    unsigned r = a.u + 0x7FFFu + ((a.u >> 16) & 1u);   // round-to-nearest-even
    union { unsigned short s; bf16 b; } o; o.s = (unsigned short)(r >> 16);
    return o.b;
}

static __device__ __forceinline__ v8f wmma_bf16(v16bf a, v16bf b, v8f c) {
    return __builtin_amdgcn_wmma_f32_16x16x32_bf16(false, a, false, b, (short)0, c, false, false);
}

// A-matrix 16x32 bf16 fragment (ISA 7.12.2): lane half 0 holds K{0..7,16..23},
// half 1 holds K{8..15,24..31}.  base = row_ptr + kstep*32 + half*8.
static __device__ __forceinline__ v16bf load_fragA(const bf16* base) {
    v8bf lo = *(const v8bf*)(base);
    v8bf hi = *(const v8bf*)(base + 16);
    v16bf r;
#pragma unroll
    for (int i = 0; i < 8; ++i) { r[i] = lo[i]; r[i + 8] = hi[i]; }
    return r;
}

// B-matrix 32x16 bf16 fragment: lanes 0-15 hold K{0..7,16..23} of their column,
// lanes 16-31 hold K{8..15,24..31}.  col0 = column ptr at K of this step.
static __device__ __forceinline__ v16bf load_fragB(const bf16* col0, int half) {
    return load_fragA(col0 + half * 8);
}

static __device__ __forceinline__ float lanexor(float v, int m, int lane) {
    int a = ((lane ^ m) << 2);
    int r = __builtin_amdgcn_ds_bpermute(a, __float_as_int(v));
    return __int_as_float(r);
}

#ifdef HAVE_TDM
typedef __attribute__((ext_vector_type(4))) unsigned v4u;
typedef __attribute__((ext_vector_type(8))) int      v8i;
typedef __attribute__((ext_vector_type(4))) int      v4i;

// TDM: DMA a 32(row) x 64(col) bf16 tile (rows contiguous, row stride 64 elems)
// from global kt into LDS.  D# packed per ISA 08 §8.3/8.4 (2D tensor, groups 2/3 zero).
// 6-arg builtin form (clang-23 / therock-10.0 toolchain).
static __device__ __forceinline__ void tdm_load_k32x64(const bf16* gptr, bf16* ldst) {
    unsigned lds_addr = (unsigned)(unsigned long long)(uintptr_t)ldst; // addr[31:0] = LDS byte offset
    unsigned long long ga = (unsigned long long)(uintptr_t)gptr;
    v4u g0;
    g0[0] = 1u;                                      // count=1, user descriptor
    g0[1] = lds_addr;                                // lds_addr
    g0[2] = (unsigned)(ga & 0xffffffffu);            // global_addr[31:0]
    g0[3] = (unsigned)((ga >> 32) & 0x1ffffffu) | (2u << 30);  // global_addr[56:32] | type=2
    v8i g1;
    g1[0] = 0x10000;                                 // wg_mask=0, data_size=1 (2 bytes)
    g1[1] = (int)(64u << 16);                        // tensor_dim0 = 64 (lo16 in hi half)
    g1[2] = (int)((4096u & 0xffffu) << 16);          // tensor_dim0 hi | tensor_dim1 lo = 4096
    g1[3] = (int)(64u << 16);                        // tensor_dim1 hi | tile_dim0 = 64
    g1[4] = 32;                                      // tile_dim1 = 32, tile_dim2 = 0
    g1[5] = 64;                                      // tensor_dim0_stride[31:0] = 64
    g1[6] = 0;                                       // stride0 hi | stride1 lo
    g1[7] = 0;                                       // stride1 hi
    v4i gz4 = {0, 0, 0, 0};
    v8i gz8 = {0, 0, 0, 0, 0, 0, 0, 0};
    __builtin_amdgcn_tensor_load_to_lds(g0, g1, gz4, gz4, gz8, 0);
}
#endif

// ---------------------------------------------------------------- kernel 0a
__global__ void pack_w_kernel(const float* __restrict__ Wq, const float* __restrict__ bq,
                              const float* __restrict__ Wk, const float* __restrict__ bk,
                              const float* __restrict__ Wv, const float* __restrict__ bv,
                              bf16* __restrict__ Wqkv, float* __restrict__ bqkv) {
    int t = blockIdx.x * 256 + threadIdx.x;
    if (t < MTOT * CC) {
        int m = t / CC, c = t % CC;
        float w = (m < CRED)     ? Wq[m * CC + c]
                : (m < 2 * CRED) ? Wk[(m - CRED) * CC + c]
                                 : Wv[(m - 2 * CRED) * CC + c];
        Wqkv[t] = f2bf(w);
    }
    if (t < MTOT) {
        bqkv[t] = (t < CRED) ? bq[t] : (t < 2 * CRED) ? bk[t - CRED] : bv[t - 2 * CRED];
    }
}

// ---------------------------------------------------------------- kernel 0b
// x (B,C,N) f32  ->  xt (B,N,C) bf16  via LDS 32x32 tile transpose
__global__ void transpose_x_kernel(const float* __restrict__ x, bf16* __restrict__ xt) {
    __shared__ float tile[32][33];
    int b  = blockIdx.z;
    int n0 = blockIdx.x * 32;
    int c0 = blockIdx.y * 32;
    int tx = threadIdx.x, ty = threadIdx.y;          // (32, 8)
    const float* xp = x + (size_t)b * CC * NTOT;
#pragma unroll
    for (int k = 0; k < 4; ++k)
        tile[ty + k * 8][tx] = xp[(size_t)(c0 + ty + k * 8) * NTOT + n0 + tx];
    __syncthreads();
    bf16* xo = xt + (size_t)b * NTOT * CC;
#pragma unroll
    for (int k = 0; k < 4; ++k)
        xo[(size_t)(n0 + ty + k * 8) * CC + c0 + tx] = f2bf(tile[tx][ty + k * 8]);
}

// ---------------------------------------------------------------- kernel 1
// Out[m,n] = sum_c Wqkv[m,c]*xt[b,n,c] + bias;  one 16x16 tile per wave.
// m<64 -> qt (B,N,64); m<128 -> kt (B,N,64); else -> vb (B,C,N)
__global__ void __launch_bounds__(256) qkv_gemm_kernel(
        const bf16* __restrict__ Wqkv, const float* __restrict__ bqkv,
        const bf16* __restrict__ xt,
        bf16* __restrict__ qt, bf16* __restrict__ kt, bf16* __restrict__ vb) {
    const int lane = threadIdx.x & 31;
    const int wave = threadIdx.x >> 5;
    const int lm   = lane & 15;
    const int half = lane >> 4;

    const int MT = MTOT / 16;      // 40
    const int NT = NTOT / 16;      // 256
    int tile = blockIdx.x * 8 + wave;
    int b  = tile / (MT * NT);
    int r0 = tile % (MT * NT);
    int m0 = (r0 % MT) * 16;
    int n0 = (r0 / MT) * 16;

    const bf16* arow = Wqkv + (size_t)(m0 + lm) * CC;
    const bf16* bcol = xt + ((size_t)b * NTOT + n0 + lm) * CC;

    v8f acc = {};
#pragma unroll 4
    for (int c = 0; c < CC; c += 32) {
        v16bf a  = load_fragA(arow + c + half * 8);
        v16bf bb = load_fragB(bcol + c, half);
        acc = wmma_bf16(a, bb, acc);
    }

    int n = n0 + lm;                                   // lane's output column
    if (m0 < 2 * CRED) {                               // q or k path, m contiguous
        v8bf outv;
#pragma unroll
        for (int r = 0; r < 8; ++r)
            outv[r] = f2bf(acc[r] + bqkv[m0 + half * 8 + r]);
        bf16* dst = (m0 < CRED)
            ? (qt + ((size_t)b * NTOT + n) * CRED + m0 + half * 8)
            : (kt + ((size_t)b * NTOT + n) * CRED + (m0 - CRED) + half * 8);
        *(v8bf*)dst = outv;
    } else {                                           // v path, strided over m
        int mb = m0 - 2 * CRED;
#pragma unroll
        for (int r = 0; r < 8; ++r) {
            int m = mb + half * 8 + r;
            vb[((size_t)b * CC + m) * NTOT + n] = f2bf(acc[r] + bqkv[m0 + half * 8 + r]);
        }
    }
}

// ---------------------------------------------------------------- kernel 2
// Flash attention: block = (b, 16-row i tile); 8 waves x 64 channels each.
// K tiles stream in via the Tensor Data Mover (double-buffered LDS);
// V columns are per-wave unique -> direct global b128 loads + prefetch.
__global__ void __launch_bounds__(256) attn_kernel(
        const bf16* __restrict__ qt, const bf16* __restrict__ kt,
        const bf16* __restrict__ vb, const float* __restrict__ x,
        const float* __restrict__ gamma, float* __restrict__ out) {
    __shared__ bf16 kbuf[2][32 * 64];                  // double-buffered K tile (TDM dest)
    __shared__ bf16 plds[8 * 512];                     // per-wave 16x32 p tile
    const int lane = threadIdx.x & 31;
    const int wave = threadIdx.x >> 5;
    const int lm   = lane & 15;
    const int half = lane >> 4;
    const int b  = blockIdx.y;
    const int i0 = blockIdx.x * 16;
    const int c0 = wave * 64;

    // q A-fragments for the whole i-tile (K = 64 -> two k-steps), kept in VGPRs
    const bf16* qrow = qt + ((size_t)b * NTOT + i0 + lm) * CRED;
    v16bf aq0 = load_fragA(qrow + 0  + half * 8);
    v16bf aq1 = load_fragA(qrow + 32 + half * 8);

    v8f acc[4] = {};
    float m_run[8], l_run[8];
#pragma unroll
    for (int r = 0; r < 8; ++r) { m_run[r] = -1e30f; l_run[r] = 0.f; }

    bf16*       pw = plds + wave * 512;
    const bf16* pr = plds + wave * 512 + lm * 32 + half * 8;
    const size_t vstride = (size_t)16 * NTOT;
    const bf16* ktb = kt + (size_t)b * NTOT * CRED;

#ifdef HAVE_TDM
    if (wave == 0) {
        tdm_load_k32x64(ktb, &kbuf[0][0]);             // preload first K tile
        __builtin_amdgcn_s_wait_tensorcnt(0);
    }
    __syncthreads();
#endif

    for (int j0 = 0, it = 0; j0 < NTOT; j0 += 32, ++it) {
        int cur = it & 1;
#ifdef HAVE_TDM
        if (wave == 0 && j0 + 32 < NTOT)               // kick off next tile DMA
            tdm_load_k32x64(ktb + (size_t)(j0 + 32) * CRED, &kbuf[cur ^ 1][0]);
        const bf16* k0 = &kbuf[cur][(0  + lm) * 64];   // col j0+lm,    K contiguous
        const bf16* k1 = &kbuf[cur][(16 + lm) * 64];   // col j0+16+lm
#else
        (void)cur;
        const bf16* k0 = ktb + (size_t)(j0 + lm) * CRED;
        const bf16* k1 = ktb + (size_t)(j0 + 16 + lm) * CRED;
#endif
        // ---- energy tile 16x32 = qT(16x64) * k(64x32), f32 accumulate ----
        v8f e0 = {}, e1 = {};
        e0 = wmma_bf16(aq0, load_fragB(k0 + 0,  half), e0);
        e0 = wmma_bf16(aq1, load_fragB(k0 + 32, half), e0);
        e1 = wmma_bf16(aq0, load_fragB(k1 + 0,  half), e1);
        e1 = wmma_bf16(aq1, load_fragB(k1 + 32, half), e1);

        // ---- online softmax (rows live in VGPR index; cols across half-wave) --
#pragma unroll
        for (int r = 0; r < 8; ++r) {
            float t = fmaxf(e0[r], e1[r]);
#pragma unroll
            for (int mm = 1; mm <= 8; mm <<= 1) t = fmaxf(t, lanexor(t, mm, lane));
            float mnew = fmaxf(m_run[r], t);
            float sc   = __expf(m_run[r] - mnew);
            m_run[r] = mnew;
            float p0 = __expf(e0[r] - mnew);
            float p1 = __expf(e1[r] - mnew);
            e0[r] = p0; e1[r] = p1;
            float s = p0 + p1;
#pragma unroll
            for (int mm = 1; mm <= 8; mm <<= 1) s += lanexor(s, mm, lane);
            l_run[r] = l_run[r] * sc + s;
#pragma unroll
            for (int ct = 0; ct < 4; ++ct) acc[ct][r] *= sc;
        }

        // ---- stage p tile (D-layout) into LDS as bf16, row-major 16x32 ----
#pragma unroll
        for (int r = 0; r < 8; ++r) {
            int row = r + half * 8;
            pw[row * 32 + lm]      = f2bf(e0[r]);
            pw[row * 32 + 16 + lm] = f2bf(e1[r]);
        }

        // ---- re-read p as A-fragment, accumulate P(16x32) * V^T(32x16) x4 ----
        v16bf ap = load_fragA(pr);
        const bf16* vcol = vb + ((size_t)b * CC + c0 + lm) * NTOT + j0;
#pragma unroll
        for (int ct = 0; ct < 4; ++ct) {
            if (j0 + 32 < NTOT)                        // prefetch next j-block of V
                __builtin_prefetch(vcol + ct * vstride + 32, 0, 1);
            acc[ct] = wmma_bf16(ap, load_fragB(vcol + ct * vstride, half), acc[ct]);
        }

#ifdef HAVE_TDM
        if (wave == 0) __builtin_amdgcn_s_wait_tensorcnt(0);
        __syncthreads();                               // publish next K tile to all waves
#endif
    }

    // ---- epilogue: normalize, gamma * out + x ----
    float g = gamma[0];
#pragma unroll
    for (int ct = 0; ct < 4; ++ct) {
        int c = c0 + ct * 16 + lm;
#pragma unroll
        for (int r = 0; r < 8; ++r) {
            int i = i0 + r + half * 8;
            size_t idx = ((size_t)b * CC + c) * NTOT + i;
            out[idx] = g * (acc[ct][r] / l_run[r]) + x[idx];
        }
    }
}

// ---------------------------------------------------------------- launcher
extern "C" void kernel_launch(void* const* d_in, const int* in_sizes, int n_in,
                              void* d_out, int out_size, void* d_ws, size_t ws_size,
                              hipStream_t stream) {
    const float* x     = (const float*)d_in[0];
    const float* Wq    = (const float*)d_in[1];
    const float* bq    = (const float*)d_in[2];
    const float* Wk    = (const float*)d_in[3];
    const float* bk    = (const float*)d_in[4];
    const float* Wv    = (const float*)d_in[5];
    const float* bv    = (const float*)d_in[6];
    const float* gamma = (const float*)d_in[7];
    float* out = (float*)d_out;

    char* ws = (char*)d_ws;
    bf16*  xt   = (bf16*)(ws);                          // 4*4096*512*2   = 16,777,216
    bf16*  Wqkv = (bf16*)(ws + 16777216);               // 640*512*2      =    655,360
    float* bqkv = (float*)(ws + 17432576);              // 640*4          =      2,560
    bf16*  qt   = (bf16*)(ws + 17435136);               // 4*4096*64*2    =  2,097,152
    bf16*  kt   = (bf16*)(ws + 19532288);               // 4*4096*64*2    =  2,097,152
    bf16*  vb   = (bf16*)(ws + 21629440);               // 4*512*4096*2   = 16,777,216
                                                        // total ~38.4 MB

    pack_w_kernel<<<(MTOT * CC + 255) / 256, 256, 0, stream>>>(Wq, bq, Wk, bk, Wv, bv, Wqkv, bqkv);

    dim3 tgrid(NTOT / 32, CC / 32, 4), tblk(32, 8);
    transpose_x_kernel<<<tgrid, tblk, 0, stream>>>(x, xt);

    int qkv_tiles = 4 * (MTOT / 16) * (NTOT / 16);      // 40960 tiles, 8 per block
    qkv_gemm_kernel<<<qkv_tiles / 8, 256, 0, stream>>>(Wqkv, bqkv, xt, qt, kt, vb);

    dim3 agrid(NTOT / 16, 4);
    attn_kernel<<<agrid, 256, 0, stream>>>(qt, kt, vb, x, gamma, out);
}
